// CollapsePredictor_53489522705077
// MI455X (gfx1250) — matrix operand order repaired and checked by reading, not compiled, not run on hardware
//
#include <hip/hip_runtime.h>
#include <hip/hip_bf16.h>
#include <math.h>

typedef _Float16 h16;
typedef __attribute__((ext_vector_type(16))) _Float16 v16h;
typedef __attribute__((ext_vector_type(8)))  float    v8f;

union UV16 { v16h v; uint4 q[2]; h16 h[16]; };
union UV8  { v8f  v; float f[8]; };

__device__ __forceinline__ v8f wmma_f16(v16h a, v16h b, v8f c) {
  // 8 args: (neg_a, A, neg_b, B, c_mod, C, reuse_a, reuse_b)
  return __builtin_amdgcn_wmma_f32_16x16x32_f16(false, a, false, b, (short)0, c, false, false);
}

// A fragment (16x32 f16, MxK) from an LDS row-major tile.
// ISA layout: lanes 0-15 hold row M=lane, K = kc*32 + {0..7, 16..23};
//             lanes 16-31 hold row M=lane-16, K = kc*32 + {8..15, 24..31}.
__device__ __forceinline__ v16h lds_a_frag(const h16* tile, int rowStride, int kc) {
  const int lane = threadIdx.x & 31;
  const int r = lane & 15;
  const h16* p = tile + r * rowStride + kc * 32 + ((lane < 16) ? 0 : 8);
  UV16 u;
  u.q[0] = *(const uint4*)(p);        // 8 halves
  u.q[1] = *(const uint4*)(p + 16);   // 8 halves, +16 element offset
  return u.v;
}

// B fragment from pre-packed global weights: [nt][kc][lane][16] halves,
// 32 bytes/lane, 32B aligned -> contiguous b128 pair (L2-resident weights).
__device__ __forceinline__ v16h glb_b_frag(const h16* __restrict__ wp, int nt, int kc, int KC) {
  const int lane = threadIdx.x & 31;
  return *(const v16h*)(wp + (((size_t)(nt * KC + kc) * 32 + lane) << 4));
}

// ---------------------------------------------------------------------------
// Pack a row-major f32 weight [Klog x 128] (per step) into WMMA B-fragment
// order, zero-padding K up to KC*32. B layout: lane<16 -> col=nt*16+lane,
// K=kc*32+{0..7,16..23}; lane>=16 -> col=nt*16+lane-16, K=kc*32+{8..15,24..31}.
// ---------------------------------------------------------------------------
__global__ void pack_b_kernel(const float* __restrict__ src, h16* __restrict__ dst,
                              int Klog, int KC, int steps) {
  const int idx = blockIdx.x * blockDim.x + threadIdx.x;
  const int per = 8 * KC * 32;
  if (idx >= per * steps) return;
  const int s    = idx / per;
  const int r    = idx - s * per;
  const int lane = r & 31;
  const int kc   = (r >> 5) % KC;
  const int nt   = r / (KC * 32);
  const int col  = nt * 16 + (lane & 15);
  const int kb   = kc * 32 + ((lane < 16) ? 0 : 8);
  const float* sp = src + (size_t)s * Klog * 128;
  h16* op = dst + (size_t)idx * 16;
#pragma unroll
  for (int i = 0; i < 16; i++) {
    const int k = kb + i + ((i < 8) ? 0 : 8);
    op[i] = (h16)((k < Klog) ? sp[(size_t)k * 128 + col] : 0.f);
  }
}

// ---------------------------------------------------------------------------
// h = node_feats @ W_in + b_in  (N x 8 x 128, trivially VALU-bound)
// ---------------------------------------------------------------------------
__global__ void input_proj_kernel(const float* __restrict__ nf, const float* __restrict__ W,
                                  const float* __restrict__ b,
                                  float* __restrict__ h32, h16* __restrict__ hf, int Nn) {
  const size_t idx = (size_t)blockIdx.x * blockDim.x + threadIdx.x;
  if (idx >= (size_t)Nn * 128) return;
  const int n = (int)(idx >> 7), j = (int)(idx & 127);
  float acc = b[j];
#pragma unroll
  for (int k = 0; k < 8; k++) acc += nf[(size_t)n * 8 + k] * W[k * 128 + j];
  h32[idx] = acc;
  hf[idx] = (h16)acc;
}

__global__ void zero_kernel(float* __restrict__ p, size_t n) {
  const size_t i = (size_t)blockIdx.x * blockDim.x + threadIdx.x;
  if (i < n) p[i] = 0.f;
}

// ---------------------------------------------------------------------------
// Edge message MLP: per 64-edge block, gather [h_src | h_dst | ef] (K=288 pad)
// into LDS, two WMMA GEMM layers (kc-outer so each A fragment is loaded once
// and feeds 4 back-to-back WMMAs), scatter-add into agg via f32 atomics.
// ---------------------------------------------------------------------------
__launch_bounds__(256)
__global__ void msg_kernel(const int* __restrict__ ei, const float* __restrict__ ef,
                           const h16* __restrict__ hf,
                           const h16* __restrict__ Wm1p, const float* __restrict__ bm1,
                           const h16* __restrict__ Wm2p, const float* __restrict__ bm2,
                           float* __restrict__ agg, int Ee) {
  __shared__ __align__(32) h16 At[4][16][288];   // 4 subtiles x 16 edges x (2H+4 pad 288)
  __shared__ __align__(32) h16 Mid[4][16][128];  // relu(layer1) in f16
  __shared__ int dstI[64];

  const int t = threadIdx.x;
  {
    const int el = t >> 2;               // edge within block (0..63)
    const int q  = t & 3;                // 64-col segment
    const int e  = blockIdx.x * 64 + el;
    const int es = el >> 4, er = el & 15;
    uint4* d4 = (uint4*)&At[es][er][q * 64];
    if (e < Ee) {
      const int sIdx = ei[e];
      const int dIdx = ei[Ee + e];
      const h16* sp = (q < 2) ? (hf + (size_t)sIdx * 128 + q * 64)
                              : (hf + (size_t)dIdx * 128 + (q - 2) * 64);
      const uint4* s4 = (const uint4*)sp;
#pragma unroll
      for (int i = 0; i < 8; i++) d4[i] = s4[i];
      if (q == 0) {
        dstI[el] = dIdx;
#pragma unroll
        for (int i = 0; i < 4; i++) At[es][er][256 + i] = (h16)ef[(size_t)e * 4 + i];
#pragma unroll
        for (int i = 4; i < 32; i++) At[es][er][256 + i] = (h16)0.f;
      }
    } else {
      uint4 z = make_uint4(0u, 0u, 0u, 0u);
#pragma unroll
      for (int i = 0; i < 8; i++) d4[i] = z;
      if (q == 0) {
        dstI[el] = -1;
        for (int i = 0; i < 32; i++) At[es][er][256 + i] = (h16)0.f;
      }
    }
  }
  __syncthreads();

  const int lane = t & 31;
  const int wave = t >> 5;
  const int es = wave >> 1;   // edge subtile
  const int nh = wave & 1;    // which half of the 8 N-tiles
  const int cl = lane & 15;

  // Layer 1: [16,288] x [288,128] -> relu -> Mid (f16)
  {
    const v8f zero = {};
    v8f acc[4] = {zero, zero, zero, zero};
#pragma unroll
    for (int kc = 0; kc < 9; kc++) {
      const v16h a = lds_a_frag(&At[es][0][0], 288, kc);
#pragma unroll
      for (int i = 0; i < 4; i++)
        acc[i] = wmma_f16(a, glb_b_frag(Wm1p, nh * 4 + i, kc, 9), acc[i]);
    }
#pragma unroll
    for (int i = 0; i < 4; i++) {
      const int ch = (nh * 4 + i) * 16 + cl;
      const float bb = bm1[ch];
      UV8 u; u.v = acc[i];
#pragma unroll
      for (int v = 0; v < 8; v++) {
        float x = u.f[v] + bb;
        x = x > 0.f ? x : 0.f;
        const int row = (lane < 16) ? v : v + 8;
        Mid[es][row][ch] = (h16)x;
      }
    }
  }
  __syncthreads();

  // Layer 2: [16,128] x [128,128] -> scatter-add agg[dst]
  {
    const v8f zero = {};
    v8f acc[4] = {zero, zero, zero, zero};
#pragma unroll
    for (int kc = 0; kc < 4; kc++) {
      const v16h a = lds_a_frag(&Mid[es][0][0], 128, kc);
#pragma unroll
      for (int i = 0; i < 4; i++)
        acc[i] = wmma_f16(a, glb_b_frag(Wm2p, nh * 4 + i, kc, 4), acc[i]);
    }
#pragma unroll
    for (int i = 0; i < 4; i++) {
      const int ch = (nh * 4 + i) * 16 + cl;
      const float bb = bm2[ch];
      UV8 u; u.v = acc[i];
#pragma unroll
      for (int v = 0; v < 8; v++) {
        const int row = (lane < 16) ? v : v + 8;
        const int d = dstI[es * 16 + row];
        if (d >= 0) atomicAdd(&agg[(size_t)d * 128 + ch], u.f[v] + bb);
      }
    }
  }
}

// ---------------------------------------------------------------------------
// GRU-style gate + LayerNorm: per 64-node block. A = [h | agg] (K=256),
// three WMMA GEMMs (z, r, h_new), elementwise blend, LN, write h32 + h16.
// ---------------------------------------------------------------------------
__launch_bounds__(256)
__global__ void gate_kernel(float* __restrict__ h32, h16* __restrict__ hf,
                            const float* __restrict__ agg,
                            const h16* __restrict__ Wzp, const float* __restrict__ bz,
                            const h16* __restrict__ Wrp, const float* __restrict__ br,
                            const h16* __restrict__ Whp, const float* __restrict__ bh,
                            const float* __restrict__ lns, const float* __restrict__ lnb,
                            int Nn) {
  __shared__ __align__(32) h16 Ag[4][16][256];  // [h | agg], h later replaced by r*h
  __shared__ float Out[4][16][128];             // pre-LN GRU output

  const int t = threadIdx.x;
  const int nodeBase = blockIdx.x * 64;
  {
    const int nl = t >> 2, q = t & 3;
    const int node = nodeBase + nl;
    const int ns = nl >> 4, nr = nl & 15;
    h16* dp = &Ag[ns][nr][q * 64];
    if (node < Nn) {
      if (q < 2) {
        const uint4* s4 = (const uint4*)(hf + (size_t)node * 128 + q * 64);
        uint4* d4 = (uint4*)dp;
#pragma unroll
        for (int i = 0; i < 8; i++) d4[i] = s4[i];
      } else {
        const float* ap = agg + (size_t)node * 128 + (q - 2) * 64;
        for (int i = 0; i < 64; i++) dp[i] = (h16)ap[i];
      }
    } else {
      for (int i = 0; i < 64; i++) dp[i] = (h16)0.f;
    }
  }
  __syncthreads();

  const int lane = t & 31;
  const int wave = t >> 5;
  const int ws = wave >> 1;
  const int nh = wave & 1;
  const int cl = lane & 15;

  float zf[4][8], rf[4][8], hv[4][8];

  {                                        // z gate
    const v8f zero = {};
    v8f acc[4] = {zero, zero, zero, zero};
#pragma unroll
    for (int kc = 0; kc < 8; kc++) {
      const v16h a = lds_a_frag(&Ag[ws][0][0], 256, kc);
#pragma unroll
      for (int i = 0; i < 4; i++)
        acc[i] = wmma_f16(a, glb_b_frag(Wzp, nh * 4 + i, kc, 8), acc[i]);
    }
#pragma unroll
    for (int i = 0; i < 4; i++) {
      const float bb = bz[(nh * 4 + i) * 16 + cl];
      UV8 u; u.v = acc[i];
#pragma unroll
      for (int v = 0; v < 8; v++) zf[i][v] = 1.f / (1.f + expf(-(u.f[v] + bb)));
    }
  }
  {                                        // r gate
    const v8f zero = {};
    v8f acc[4] = {zero, zero, zero, zero};
#pragma unroll
    for (int kc = 0; kc < 8; kc++) {
      const v16h a = lds_a_frag(&Ag[ws][0][0], 256, kc);
#pragma unroll
      for (int i = 0; i < 4; i++)
        acc[i] = wmma_f16(a, glb_b_frag(Wrp, nh * 4 + i, kc, 8), acc[i]);
    }
#pragma unroll
    for (int i = 0; i < 4; i++) {
      const float bb = br[(nh * 4 + i) * 16 + cl];
      UV8 u; u.v = acc[i];
#pragma unroll
      for (int v = 0; v < 8; v++) rf[i][v] = 1.f / (1.f + expf(-(u.f[v] + bb)));
    }
  }
  // full-precision h values for the blend / r*h
#pragma unroll
  for (int i = 0; i < 4; i++) {
    const int ch = (nh * 4 + i) * 16 + cl;
#pragma unroll
    for (int v = 0; v < 8; v++) {
      const int row = (lane < 16) ? v : v + 8;
      const int node = nodeBase + ws * 16 + row;
      hv[i][v] = (node < Nn) ? h32[(size_t)node * 128 + ch] : 0.f;
    }
  }
  __syncthreads();  // everyone finished reading h columns of Ag
#pragma unroll
  for (int i = 0; i < 4; i++) {          // overwrite h cols with r*h
    const int ch = (nh * 4 + i) * 16 + cl;
#pragma unroll
    for (int v = 0; v < 8; v++) {
      const int row = (lane < 16) ? v : v + 8;
      Ag[ws][row][ch] = (h16)(rf[i][v] * hv[i][v]);
    }
  }
  __syncthreads();
  {                                        // h_new
    const v8f zero = {};
    v8f acc[4] = {zero, zero, zero, zero};
#pragma unroll
    for (int kc = 0; kc < 8; kc++) {
      const v16h a = lds_a_frag(&Ag[ws][0][0], 256, kc);
#pragma unroll
      for (int i = 0; i < 4; i++)
        acc[i] = wmma_f16(a, glb_b_frag(Whp, nh * 4 + i, kc, 8), acc[i]);
    }
#pragma unroll
    for (int i = 0; i < 4; i++) {
      const int ch = (nh * 4 + i) * 16 + cl;
      const float bb = bh[ch];
      UV8 u; u.v = acc[i];
#pragma unroll
      for (int v = 0; v < 8; v++) {
        const float hn = tanhf(u.f[v] + bb);
        const float z = zf[i][v];
        const int row = (lane < 16) ? v : v + 8;
        Out[ws][row][ch] = (1.f - z) * hv[i][v] + z * hn;
      }
    }
  }
  __syncthreads();

  if (t < 64) {                          // LayerNorm, one thread per node row
    const int node = nodeBase + t;
    if (node < Nn) {
      const int ns = t >> 4, nr = t & 15;
      float sum = 0.f, sq = 0.f;
      for (int c = 0; c < 128; c++) { const float x = Out[ns][nr][c]; sum += x; sq += x * x; }
      const float m = sum * (1.f / 128.f);
      const float var = sq * (1.f / 128.f) - m * m;
      const float rs = rsqrtf(var + 1e-6f);
      for (int c = 0; c < 128; c++) {
        const float x = (Out[ns][nr][c] - m) * rs * lns[c] + lnb[c];
        h32[(size_t)node * 128 + c] = x;
        hf[(size_t)node * 128 + c] = (h16)x;
      }
    }
  }
}

// ---------------------------------------------------------------------------
// Heads: cls = relu(h@Wc1+b)@Wc2+b ; prob = sigmoid(relu(h@Wp1+b)@Wp2+b)
// ---------------------------------------------------------------------------
__launch_bounds__(128)
__global__ void head_kernel(const float* __restrict__ h32,
                            const float* __restrict__ Wc1, const float* __restrict__ bc1,
                            const float* __restrict__ Wc2, const float* __restrict__ bc2,
                            const float* __restrict__ Wp1, const float* __restrict__ bp1,
                            const float* __restrict__ Wp2, const float* __restrict__ bp2,
                            float* __restrict__ out, int Nn) {
  __shared__ float Hs[8][128];
  __shared__ float Hc[8][64];
  __shared__ float Hp[8][64];
  const int t = threadIdx.x;
  const int node0 = blockIdx.x * 8;
  for (int i = t; i < 8 * 128; i += 128) {
    const int nn = i >> 7, c = i & 127;
    const int node = node0 + nn;
    Hs[nn][c] = (node < Nn) ? h32[(size_t)node * 128 + c] : 0.f;
  }
  __syncthreads();
  for (int i = t; i < 8 * 64; i += 128) {
    const int nn = i >> 6, j = i & 63;
    float a1 = bc1[j], a2 = bp1[j];
    for (int k = 0; k < 128; k++) {
      const float hvv = Hs[nn][k];
      a1 += hvv * Wc1[k * 64 + j];
      a2 += hvv * Wp1[k * 64 + j];
    }
    Hc[nn][j] = fmaxf(a1, 0.f);
    Hp[nn][j] = fmaxf(a2, 0.f);
  }
  __syncthreads();
  for (int i = t; i < 48; i += 128) {
    const int nn = i / 6, c = i % 6;
    const int node = node0 + nn;
    if (node >= Nn) continue;
    if (c < 5) {
      float a = bc2[c];
      for (int k = 0; k < 64; k++) a += Hc[nn][k] * Wc2[k * 5 + c];
      out[(size_t)node * 5 + c] = a;
    } else {
      float a = bp2[0];
      for (int k = 0; k < 64; k++) a += Hp[nn][k] * Wp2[k];
      out[(size_t)Nn * 5 + node] = 1.f / (1.f + expf(-a));
    }
  }
}

// ---------------------------------------------------------------------------
extern "C" void kernel_launch(void* const* d_in, const int* in_sizes, int n_in,
                              void* d_out, int out_size, void* d_ws, size_t ws_size,
                              hipStream_t stream) {
  const float* node_feats = (const float*)d_in[0];
  const int*   edge_index = (const int*)d_in[1];
  const float* edge_feats = (const float*)d_in[2];
  const float* W_in = (const float*)d_in[3];
  const float* b_in = (const float*)d_in[4];
  const float* W_m1 = (const float*)d_in[5];
  const float* b_m1 = (const float*)d_in[6];
  const float* W_m2 = (const float*)d_in[7];
  const float* b_m2 = (const float*)d_in[8];
  const float* W_z  = (const float*)d_in[9];
  const float* b_z  = (const float*)d_in[10];
  const float* W_r  = (const float*)d_in[11];
  const float* b_r  = (const float*)d_in[12];
  const float* W_h  = (const float*)d_in[13];
  const float* b_h  = (const float*)d_in[14];
  const float* ln_s = (const float*)d_in[15];
  const float* ln_b = (const float*)d_in[16];
  const float* W_c1 = (const float*)d_in[17];
  const float* b_c1 = (const float*)d_in[18];
  const float* W_c2 = (const float*)d_in[19];
  const float* b_c2 = (const float*)d_in[20];
  const float* W_p1 = (const float*)d_in[21];
  const float* b_p1 = (const float*)d_in[22];
  const float* W_p2 = (const float*)d_in[23];
  const float* b_p2 = (const float*)d_in[24];

  const int Nn = in_sizes[0] / 8;   // 100,000
  const int Ee = in_sizes[1] / 2;   // 1,600,000
  const int S = 6;

  // Workspace carve (assumes ws_size >= ~131 MB, all offsets 256B aligned)
  char* ws = (char*)d_ws;
  size_t off = 0;
  float* h32 = (float*)(ws + off); off += (size_t)Nn * 128 * 4;
  h16*   hf  = (h16*)(ws + off);   off += (size_t)Nn * 128 * 2;
  float* agg = (float*)(ws + off); off += (size_t)Nn * 128 * 4;
  h16* Wm1p = (h16*)(ws + off); off += (size_t)S * 8 * 9 * 512 * 2;
  h16* Wm2p = (h16*)(ws + off); off += (size_t)S * 8 * 4 * 512 * 2;
  h16* Wzp  = (h16*)(ws + off); off += (size_t)S * 8 * 8 * 512 * 2;
  h16* Wrp  = (h16*)(ws + off); off += (size_t)S * 8 * 8 * 512 * 2;
  h16* Whp  = (h16*)(ws + off); off += (size_t)S * 8 * 8 * 512 * 2;
  (void)n_in; (void)out_size; (void)ws_size; (void)off;

  // Pack all per-step weights into WMMA B-fragment layout (f16)
  {
    int th;
    th = S * 8 * 9 * 32; pack_b_kernel<<<(th + 255) / 256, 256, 0, stream>>>(W_m1, Wm1p, 260, 9, S);
    th = S * 8 * 4 * 32; pack_b_kernel<<<(th + 255) / 256, 256, 0, stream>>>(W_m2, Wm2p, 128, 4, S);
    th = S * 8 * 8 * 32;
    pack_b_kernel<<<(th + 255) / 256, 256, 0, stream>>>(W_z, Wzp, 256, 8, S);
    pack_b_kernel<<<(th + 255) / 256, 256, 0, stream>>>(W_r, Wrp, 256, 8, S);
    pack_b_kernel<<<(th + 255) / 256, 256, 0, stream>>>(W_h, Whp, 256, 8, S);
  }

  const unsigned nElemBlocks = (unsigned)(((size_t)Nn * 128 + 255) / 256);
  input_proj_kernel<<<nElemBlocks, 256, 0, stream>>>(node_feats, W_in, b_in, h32, hf, Nn);

  for (int s = 0; s < S; s++) {
    zero_kernel<<<nElemBlocks, 256, 0, stream>>>(agg, (size_t)Nn * 128);
    msg_kernel<<<(Ee + 63) / 64, 256, 0, stream>>>(
        edge_index, edge_feats, hf,
        Wm1p + (size_t)s * 8 * 9 * 512, b_m1 + s * 128,
        Wm2p + (size_t)s * 8 * 4 * 512, b_m2 + s * 128,
        agg, Ee);
    gate_kernel<<<(Nn + 63) / 64, 256, 0, stream>>>(
        h32, hf, agg,
        Wzp + (size_t)s * 8 * 8 * 512, b_z + s * 128,
        Wrp + (size_t)s * 8 * 8 * 512, b_r + s * 128,
        Whp + (size_t)s * 8 * 8 * 512, b_h + s * 128,
        ln_s + s * 128, ln_b + s * 128, Nn);
  }

  head_kernel<<<(Nn + 7) / 8, 128, 0, stream>>>(
      h32, W_c1, b_c1, W_c2, b_c2, W_p1, b_p1, W_p2, b_p2, (float*)d_out, Nn);
}
